// ProMambaBlock_62302795595838
// MI455X (gfx1250) — matrix operand order repaired and checked by reading, not compiled, not run on hardware
//
#include <hip/hip_runtime.h>
#include <math.h>

// ---------------------------------------------------------------------------
// Types for CDNA5 WMMA (wave32): v_wmma_f32_16x16x32_bf16
// ---------------------------------------------------------------------------
typedef __bf16 bf16_t;
typedef __attribute__((ext_vector_type(16))) __bf16 v16bf;
typedef __attribute__((ext_vector_type(8)))  float  v8f;
typedef unsigned int u32x4 __attribute__((ext_vector_type(4)));
typedef int          i32x4 __attribute__((ext_vector_type(4)));
typedef int          i32x8 __attribute__((ext_vector_type(8)));

#if defined(__has_builtin)
#if __has_builtin(__builtin_amdgcn_tensor_load_to_lds) && __has_builtin(__builtin_amdgcn_s_wait_tensorcnt)
#define HAVE_TDM 1
#else
#define HAVE_TDM 0
#endif
#else
#define HAVE_TDM 0
#endif

struct U128 { unsigned x, y, z, w; };   // trivial 16B chunk (union-safe)

__device__ __forceinline__ bf16_t f2bf(float f) { return (bf16_t)f; }  // v_cvt_*bf16_f32

__device__ __forceinline__ v8f zero8() {
    v8f v;
#pragma unroll
    for (int i = 0; i < 8; ++i) v[i] = 0.f;
    return v;
}
__device__ __forceinline__ v16bf zerofrag() {
    union { unsigned u[8]; v16bf v; } t;
#pragma unroll
    for (int i = 0; i < 8; ++i) t.u[i] = 0u;
    return t.v;
}
// Two contiguous 16-byte chunks -> one 16-element bf16 fragment.
__device__ __forceinline__ v16bf load_frag2(const bf16_t* p, int off2) {
    union { U128 u[2]; v16bf v; } t;
    t.u[0] = *reinterpret_cast<const U128*>(p);
    t.u[1] = *reinterpret_cast<const U128*>(p + off2);
    return t.v;
}

__device__ __forceinline__ float elu1(float x)   { return x > 0.f ? x + 1.f : __expf(x); }
__device__ __forceinline__ float gelu_exact(float x) { return 0.5f * x * (1.f + erff(x * 0.70710678118654752f)); }
__device__ __forceinline__ float sigmoidf_(float x) { return 1.f / (1.f + __expf(-x)); }

// ---------------------------------------------------------------------------
// Problem constants
// ---------------------------------------------------------------------------
#define BN 8
#define LN 2048
#define CN 512
#define HN 8
#define DN 64
#define FFN 2048
#define MROWS (BN * LN)     // 16384

// ---------------------------------------------------------------------------
// TDM issue: 2D tile load (tile_dim0 = 64 K-elems, tile_dim1 = 128 rows, bf16)
// D# built per CDNA5 ISA ch.8 (group0: count/lds_addr/global_addr/type=2;
// group1: data_size=2B, tensor dims, tile dims, dim0 stride).
// ---------------------------------------------------------------------------
#if HAVE_TDM
__device__ __forceinline__ void tdm_load_tile_128x64(
    const bf16_t* gsrc, unsigned lds_addr, int k_elems, int m_elems)
{
    unsigned long long ga = (unsigned long long)(size_t)gsrc;
    u32x4 g0;
    g0[0] = 1u;                                            // count=1 (valid user D#)
    g0[1] = lds_addr;                                      // LDS byte address
    g0[2] = (unsigned)(ga & 0xFFFFFFFFull);                // global_addr[31:0]
    g0[3] = (unsigned)((ga >> 32) & 0x01FFFFFFull) | (2u << 30); // [56:32] | type=2
    i32x8 g1;
    g1[0] = (int)(1u << 16);                               // data_size=1 -> 2 bytes
    g1[1] = (int)(((unsigned)k_elems & 0xFFFFu) << 16);    // tensor_dim0[15:0]
    g1[2] = (int)((((unsigned)k_elems >> 16) & 0xFFFFu) |
                  (((unsigned)m_elems & 0xFFFFu) << 16));  // dim0 hi | dim1 lo
    g1[3] = (int)((((unsigned)m_elems >> 16) & 0xFFFFu) |
                  (64u << 16));                            // dim1 hi | tile_dim0=64
    g1[4] = (int)128u;                                     // tile_dim1=128, tile_dim2=0
    g1[5] = (int)(unsigned)k_elems;                        // tensor_dim0_stride[31:0]
    g1[6] = 0;                                             // stride hi | dim1_stride lo
    g1[7] = 0;
    i32x4 z4; z4[0] = z4[1] = z4[2] = z4[3] = 0;
#if __clang_major__ >= 23
    i32x8 z8;
#pragma unroll
    for (int i = 0; i < 8; ++i) z8[i] = 0;
    __builtin_amdgcn_tensor_load_to_lds(g0, g1, z4, z4, z8, 0);
#else
    __builtin_amdgcn_tensor_load_to_lds(g0, g1, z4, z4, 0);
#endif
}
#endif

// ---------------------------------------------------------------------------
// Generic bf16 WMMA GEMM:  C[M,N] = A[M,K] @ BT[N,K]^T  (+bias, epilogue)
// Block: 256 threads = 8 waves as 2(M)x4(N); wave tile 64x64 (4x4 wmma tiles).
// USE_TDM: stage the 128x64 A tile in LDS via tensor_load_to_lds (double
// buffered; wave0 issues DMA + s_wait_tensorcnt, barrier releases consumers).
// EPI: 0 qk(elu+1 split q/k)  1 ->bf16  2 conv-first(write+bias, row shift)
//      3 conv-acc(+=, row shift)        4 gelu->bf16  5 residual += into o0
// ---------------------------------------------------------------------------
template<int EPI, bool USE_TDM>
__global__ __launch_bounds__(256) void gemm_bf16_kernel(
    const bf16_t* __restrict__ A, const bf16_t* __restrict__ BT,
    const float* __restrict__ bias, int M, int N, int K, int shift,
    float* __restrict__ o0, float* __restrict__ o1, bf16_t* __restrict__ ob)
{
    const int lane = threadIdx.x & 31;
    const int w    = threadIdx.x >> 5;
    const int hi   = lane >> 4;       // 0/1 : lane half
    const int l15  = lane & 15;
    const int m0 = blockIdx.y * 128 + (w & 1) * 64;
    const int n0 = blockIdx.x * 256 + (w >> 1) * 64;

    v8f acc[4][4];
#pragma unroll
    for (int i = 0; i < 4; ++i)
#pragma unroll
        for (int j = 0; j < 4; ++j) acc[i][j] = zero8();

#if HAVE_TDM
    if constexpr (USE_TDM) {
        __shared__ bf16_t smem[2][128 * 64];               // 2 x 16 KB (of 320 KB)
        const int NK = K >> 6;                              // 64-wide K chunks
        const int mblk = blockIdx.y * 128;
        const bool issuer = (threadIdx.x < 32);             // wave 0 drives the TDM
        if (issuer)
            tdm_load_tile_128x64(A + (size_t)mblk * K,
                                 (unsigned)(size_t)&smem[0][0], K, M);
        for (int kc = 0; kc < NK; ++kc) {
            if (issuer) {
                if (kc + 1 < NK) {
                    tdm_load_tile_128x64(A + (size_t)mblk * K + (size_t)(kc + 1) * 64,
                                         (unsigned)(size_t)&smem[(kc + 1) & 1][0], K, M);
                    __builtin_amdgcn_s_wait_tensorcnt(1);   // chunk kc landed
                } else {
                    __builtin_amdgcn_s_wait_tensorcnt(0);
                }
            }
            __syncthreads();
            const bf16_t* sA = &smem[kc & 1][0];
#pragma unroll
            for (int t2 = 0; t2 < 2; ++t2) {
                const int k0 = kc * 64 + t2 * 32;
                v16bf af[4], bfb[4];
#pragma unroll
                for (int mi = 0; mi < 4; ++mi) {
                    int rloc = (w & 1) * 64 + mi * 16 + l15;
                    af[mi] = load_frag2(sA + rloc * 64 + t2 * 32 + hi * 8, 16);
                }
#pragma unroll
                for (int ni = 0; ni < 4; ++ni) {
                    int n = n0 + ni * 16 + l15;
                    bfb[ni] = load_frag2(BT + (size_t)n * K + k0 + hi * 8, 16);
                }
#pragma unroll
                for (int mi = 0; mi < 4; ++mi)
#pragma unroll
                    for (int ni = 0; ni < 4; ++ni)
                        acc[mi][ni] = __builtin_amdgcn_wmma_f32_16x16x32_bf16(
                            false, af[mi], false, bfb[ni], (short)0, acc[mi][ni], false, false);
            }
            __syncthreads();                                // buffer reuse fence
        }
    } else
#endif
    {
        for (int k0 = 0; k0 < K; k0 += 32) {
            v16bf af[4], bfb[4];
#pragma unroll
            for (int mi = 0; mi < 4; ++mi) {
                int row = m0 + mi * 16 + l15;
                if (EPI == 2 || EPI == 3) {
                    int ls = (row & (LN - 1)) + shift;       // position within sequence
                    if ((unsigned)ls < (unsigned)LN)
                        af[mi] = load_frag2(A + (size_t)(row + shift) * K + k0 + hi * 8, 16);
                    else
                        af[mi] = zerofrag();
                } else {
                    af[mi] = load_frag2(A + (size_t)row * K + k0 + hi * 8, 16);
                }
            }
#pragma unroll
            for (int ni = 0; ni < 4; ++ni) {
                int n = n0 + ni * 16 + l15;
                bfb[ni] = load_frag2(BT + (size_t)n * K + k0 + hi * 8, 16);
            }
#pragma unroll
            for (int mi = 0; mi < 4; ++mi)
#pragma unroll
                for (int ni = 0; ni < 4; ++ni)
                    acc[mi][ni] = __builtin_amdgcn_wmma_f32_16x16x32_bf16(
                        false, af[mi], false, bfb[ni], (short)0, acc[mi][ni], false, false);
        }
    }

#pragma unroll
    for (int mi = 0; mi < 4; ++mi) {
#pragma unroll
        for (int ni = 0; ni < 4; ++ni) {
#pragma unroll
            for (int r = 0; r < 8; ++r) {
                int row = m0 + mi * 16 + r + hi * 8;
                int col = n0 + ni * 16 + l15;
                float v = acc[mi][ni][r];
                size_t idx = (size_t)row * N + col;
                if (EPI == 0) {
                    v = elu1(v + bias[col]);
                    if (col < CN) o0[(size_t)row * CN + col] = v;
                    else          o1[(size_t)row * CN + (col - CN)] = v;
                } else if (EPI == 1) {
                    ob[idx] = f2bf(v + bias[col]);
                } else if (EPI == 2) {
                    o0[idx] = v + bias[col];
                } else if (EPI == 3) {
                    o0[idx] += v;
                } else if (EPI == 4) {
                    ob[idx] = f2bf(gelu_exact(v + bias[col]));
                } else { // 5: residual accumulate
                    o0[idx] += v + bias[col];
                }
            }
        }
    }
}

// ---------------------------------------------------------------------------
// kv[b,h] = (k^T v)/L : 64x64 per (b,h). One workgroup per (b,h), 8 waves,
// each wave owns 2 of the 16 wmma tiles, full K=L sweep (strided gathers).
// Output written transposed [e][d] as bf16 so the next GEMM's B loads are
// contiguous.
// ---------------------------------------------------------------------------
__global__ __launch_bounds__(256) void kv_kernel(
    const bf16_t* __restrict__ kbf, const bf16_t* __restrict__ vbf,
    bf16_t* __restrict__ kvT)
{
    const int bh = blockIdx.x;
    const int b = bh >> 3, h = bh & 7;
    const size_t gbase = (size_t)b * LN * CN + h * DN;
    const int lane = threadIdx.x & 31;
    const int w    = threadIdx.x >> 5;
    const int hi = lane >> 4, l15 = lane & 15;

    v8f acc[2];
    acc[0] = zero8(); acc[1] = zero8();
    const int tile[2] = { w, w + 8 };

    for (int l0 = 0; l0 < LN; l0 += 32) {
#pragma unroll
        for (int t = 0; t < 2; ++t) {
            const int mi = tile[t] >> 2, ni = tile[t] & 3;
            v16bf a, bb;
#pragma unroll
            for (int j = 0; j < 16; ++j) {   // A = k^T : M=d, K=l
                int l = l0 + hi * 8 + (j & 7) + ((j >> 3) << 4);
                a[j] = kbf[gbase + (size_t)l * CN + mi * 16 + l15];
            }
#pragma unroll
            for (int j = 0; j < 16; ++j) {   // B = v   : K=l, N=e
                int l = l0 + hi * 8 + (j & 7) + ((j >> 3) << 4);
                bb[j] = vbf[gbase + (size_t)l * CN + ni * 16 + l15];
            }
            acc[t] = __builtin_amdgcn_wmma_f32_16x16x32_bf16(
                false, a, false, bb, (short)0, acc[t], false, false);
        }
    }
#pragma unroll
    for (int t = 0; t < 2; ++t) {
        const int mi = tile[t] >> 2, ni = tile[t] & 3;
#pragma unroll
        for (int r = 0; r < 8; ++r) {
            int d = mi * 16 + r + hi * 8;
            int e = ni * 16 + l15;
            kvT[((size_t)bh * DN + e) * DN + d] = f2bf(acc[t][r] * (1.0f / (float)LN));
        }
    }
}

// ---------------------------------------------------------------------------
// out[b,h,l,e] = x[b,l,h*64+e] + z[b,h,l] * (q[b,h,l,:] @ kv[b,h,:,e])
// Grid (B*H, L/512). 8 waves, each wave 64 rows x 64 cols (4x4 tiles), K=64.
// ---------------------------------------------------------------------------
__global__ __launch_bounds__(256) void attn_out_kernel(
    const bf16_t* __restrict__ qbf, const bf16_t* __restrict__ kvT,
    const float* __restrict__ z, const float* __restrict__ xin,
    float* __restrict__ out)
{
    const int bh = blockIdx.x;
    const int b = bh >> 3, h = bh & 7;
    const int lane = threadIdx.x & 31;
    const int w    = threadIdx.x >> 5;
    const int hi = lane >> 4, l15 = lane & 15;
    const int lbase = blockIdx.y * 512 + w * 64;

    v8f acc[4][4];
#pragma unroll
    for (int i = 0; i < 4; ++i)
#pragma unroll
        for (int j = 0; j < 4; ++j) acc[i][j] = zero8();

#pragma unroll
    for (int k0 = 0; k0 < DN; k0 += 32) {
        v16bf af[4], bfb[4];
#pragma unroll
        for (int mi = 0; mi < 4; ++mi) {
            int l = lbase + mi * 16 + l15;
            af[mi] = load_frag2(qbf + ((size_t)(b * LN + l)) * CN + h * DN + k0 + hi * 8, 16);
        }
#pragma unroll
        for (int ni = 0; ni < 4; ++ni) {
            bfb[ni] = load_frag2(kvT + ((size_t)bh * DN + ni * 16 + l15) * DN + k0 + hi * 8, 16);
        }
#pragma unroll
        for (int mi = 0; mi < 4; ++mi)
#pragma unroll
            for (int ni = 0; ni < 4; ++ni)
                acc[mi][ni] = __builtin_amdgcn_wmma_f32_16x16x32_bf16(
                    false, af[mi], false, bfb[ni], (short)0, acc[mi][ni], false, false);
    }
#pragma unroll
    for (int mi = 0; mi < 4; ++mi) {
#pragma unroll
        for (int ni = 0; ni < 4; ++ni) {
#pragma unroll
            for (int r = 0; r < 8; ++r) {
                int l = lbase + mi * 16 + r + hi * 8;
                int e = ni * 16 + l15;
                size_t gi = (size_t)(b * LN + l) * CN + h * DN + e;
                out[gi] = xin[gi] + z[(size_t)bh * LN + l] * acc[mi][ni][r];
            }
        }
    }
}

// ---------------------------------------------------------------------------
// LayerNorm over C=512: 8 waves / block, one wave per row, shfl_xor reduce.
// ---------------------------------------------------------------------------
__global__ __launch_bounds__(256) void ln_kernel(
    const float* __restrict__ in, const float* __restrict__ g,
    const float* __restrict__ bta, bf16_t* __restrict__ outb,
    float* __restrict__ outf)
{
    const int w = threadIdx.x >> 5;
    const int lane = threadIdx.x & 31;
    const int row = blockIdx.x * 8 + w;
    const float* p = in + (size_t)row * CN;
    float v[16], s = 0.f, ss = 0.f;
#pragma unroll
    for (int i = 0; i < 16; ++i) { v[i] = p[i * 32 + lane]; s += v[i]; ss += v[i] * v[i]; }
#pragma unroll
    for (int off = 16; off > 0; off >>= 1) { s += __shfl_xor(s, off, 32); ss += __shfl_xor(ss, off, 32); }
    const float mu = s * (1.f / CN);
    const float var = ss * (1.f / CN) - mu * mu;
    const float rs = rsqrtf(var + 1e-5f);
#pragma unroll
    for (int i = 0; i < 16; ++i) {
        int c = i * 32 + lane;
        float y = (v[i] - mu) * rs * g[c] + bta[c];
        if (outb) outb[(size_t)row * CN + c] = f2bf(y);
        if (outf) outf[(size_t)row * CN + c] = y;
    }
}

// ---------------------------------------------------------------------------
// LePE: faithful to torch raw reshape [B,H,L,D]->[B*H,D,L]; channel ch data
// at flat index f = ch*L + pos within each (b,h) block (f = l*D + d coords).
// out[b,h,l,d] = t[b,h,l,d] + sum_k w[d,k]*src(f = d*L + (l+k-1)) + bias[d]
// ---------------------------------------------------------------------------
__global__ __launch_bounds__(256) void lepe_kernel(
    const float* __restrict__ src, const float* __restrict__ wgt,
    const float* __restrict__ bias, float* __restrict__ dstf,
    bf16_t* __restrict__ dstb)
{
    int t = blockIdx.x * 256 + threadIdx.x;            // over B*H*L*D
    int bh = t >> 17;                                  // L*D = 131072
    int b = bh >> 3, h = bh & 7;
    int rest = t & 131071;
    int l = rest >> 6, d = rest & 63;
    const size_t base = (size_t)b * LN * CN + h * DN;

    float xc;
    {   // src at flat f = d*L + l  -> coords (ls = f/64, ds = f%64)
        int f = d * LN + l;
        xc = src[base + (size_t)(f >> 6) * CN + (f & 63)];
    }
    float xl = 0.f, xr = 0.f;
    if (l > 0)      { int f = d * LN + (l - 1); xl = src[base + (size_t)(f >> 6) * CN + (f & 63)]; }
    if (l < LN - 1) { int f = d * LN + (l + 1); xr = src[base + (size_t)(f >> 6) * CN + (f & 63)]; }

    float center = src[base + (size_t)l * CN + d];     // t[b,h,l,d]
    float o = center + wgt[d * 3] * xl + wgt[d * 3 + 1] * xc + wgt[d * 3 + 2] * xr + bias[d];
    size_t a = base + (size_t)l * CN + d;
    dstf[a] = o;
    dstb[a] = f2bf(o);
}

// k_mean[b, c] = mean_l k[b,l,c]   (c = h*64+d)
__global__ __launch_bounds__(256) void kmean_kernel(
    const float* __restrict__ kL, float* __restrict__ km)
{
    int t = blockIdx.x * 256 + threadIdx.x;  // over B*512
    int b = t >> 9, c = t & 511;
    float s = 0.f;
    for (int l = 0; l < LN; ++l) s += kL[((size_t)(b * LN + l)) * CN + c];
    km[t] = s * (1.f / (float)LN);
}

// z[b,h,l] = 1 / (q[b,h,l,:] . k_mean[b,h,:] + 1e-6)
__global__ __launch_bounds__(256) void z_kernel(
    const float* __restrict__ qL, const float* __restrict__ km,
    float* __restrict__ z)
{
    int t = blockIdx.x * 256 + threadIdx.x;  // over MROWS*H
    int g = t >> 3, h = t & 7;
    int b = g >> 11, l = g & (LN - 1);
    const float* qp = qL + (size_t)g * CN + h * DN;
    const float* kp = km + (size_t)b * CN + h * DN;
    float s = 0.f;
#pragma unroll 8
    for (int d = 0; d < DN; ++d) s += qp[d] * kp[d];
    z[((size_t)(b * HN + h)) * LN + l] = 1.f / (s + 1e-6f);
}

// x += sp_in + vv * sigmoid(gg)   (vv = c[:, :512], gg = c[:, 512:])
__global__ __launch_bounds__(256) void gate_kernel(
    const float* __restrict__ c, const float* __restrict__ spin,
    float* __restrict__ out)
{
    int t = blockIdx.x * 256 + threadIdx.x;   // over MROWS*512
    int row = t >> 9, col = t & 511;
    float vv = c[(size_t)row * 1024 + col];
    float gg = c[(size_t)row * 1024 + 512 + col];
    out[t] += spin[t] + vv * sigmoidf_(gg);
}

// W[K,N] fp32 -> WT[N,K] bf16
__global__ __launch_bounds__(256) void trbf_kernel(
    const float* __restrict__ in, bf16_t* __restrict__ out, int K, int N)
{
    int t = blockIdx.x * 256 + threadIdx.x;   // over N*K (output order)
    int n = t / K, k = t - n * K;
    out[t] = f2bf(in[(size_t)k * N + n]);
}

// sp_cw[co, ci, 3] fp32 -> three bf16 matrices Ws[s][co][ci]
__global__ __launch_bounds__(256) void spw_kernel(
    const float* __restrict__ in, bf16_t* __restrict__ w0,
    bf16_t* __restrict__ w1, bf16_t* __restrict__ w2)
{
    int t = blockIdx.x * 256 + threadIdx.x;   // over 1024*512
    w0[t] = f2bf(in[(size_t)t * 3 + 0]);
    w1[t] = f2bf(in[(size_t)t * 3 + 1]);
    w2[t] = f2bf(in[(size_t)t * 3 + 2]);
}

// ---------------------------------------------------------------------------
// Host launcher
// ---------------------------------------------------------------------------
extern "C" void kernel_launch(void* const* d_in, const int* in_sizes, int n_in,
                              void* d_out, int out_size, void* d_ws, size_t ws_size,
                              hipStream_t stream)
{
    const float* x      = (const float*)d_in[0];
    const float* g1     = (const float*)d_in[1];
    const float* b1     = (const float*)d_in[2];
    const float* qk_w   = (const float*)d_in[3];
    const float* qk_b   = (const float*)d_in[4];
    const float* v_w    = (const float*)d_in[5];
    const float* v_b    = (const float*)d_in[6];
    const float* lepe_w = (const float*)d_in[7];
    const float* lepe_b = (const float*)d_in[8];
    const float* g2     = (const float*)d_in[9];
    const float* b2     = (const float*)d_in[10];
    const float* spn_g  = (const float*)d_in[11];
    const float* spn_b  = (const float*)d_in[12];
    const float* sp_cw  = (const float*)d_in[13];
    const float* sp_cb  = (const float*)d_in[14];
    const float* g3     = (const float*)d_in[15];
    const float* b3     = (const float*)d_in[16];
    const float* w1     = (const float*)d_in[17];
    const float* bb1    = (const float*)d_in[18];
    const float* w2     = (const float*)d_in[19];
    const float* bb2    = (const float*)d_in[20];
    float* out = (float*)d_out;

    char* ws = (char*)d_ws;
    // Workspace layout (bytes). Total ~234 MiB with aliasing.
    const size_t OA    = 0;                          // LN-out bf16 A operand  [M,512]
    const size_t OQKT  = OA    + 16777216;           // qk_w^T bf16 [1024,512]
    const size_t OVT   = OQKT  + 1048576;            // v_w^T  bf16 [512,512]
    const size_t OSP0  = OVT   + 524288;             // conv W tap0 [1024,512]
    const size_t OSP1  = OSP0  + 1048576;
    const size_t OSP2  = OSP1  + 1048576;
    const size_t OW1T  = OSP2  + 1048576;            // w1^T [2048,512]
    const size_t OW2T  = OW1T  + 2097152;            // w2^T [512,2048]
    const size_t OQ0   = OW2T  + 2097152;            // q pre-LePE fp32 [M,512]
    const size_t OK0   = OQ0   + 33554432;           // k pre-LePE fp32
    const size_t OQL   = OK0   + 33554432;           // q post-LePE fp32
    const size_t OKL   = OQL   + 33554432;           // k post-LePE fp32
    const size_t OQB   = OKL   + 33554432;           // q bf16
    const size_t OKB   = OQB   + 16777216;           // k bf16
    const size_t OVB   = OKB   + 16777216;           // v bf16
    const size_t OKV   = OVB   + 16777216;           // kv^T bf16 [BH,64,64]
    const size_t OKM   = OKV   + 524288;             // k_mean fp32 [B,512]
    const size_t OZ    = OKM   + 16384;              // z fp32 [BH,L]
    const size_t OSPIN = OZ    + 524288;             // sp_in fp32 [M,512]
    const size_t OC    = OQ0;                        // conv out fp32 [M,1024] (aliases q0/k0, dead)
    const size_t OU    = OQL;                        // FFN hidden bf16 [M,2048] (aliases qL/kL, dead)

    bf16_t* Abf  = (bf16_t*)(ws + OA);
    bf16_t* qkT  = (bf16_t*)(ws + OQKT);
    bf16_t* vT   = (bf16_t*)(ws + OVT);
    bf16_t* sp0  = (bf16_t*)(ws + OSP0);
    bf16_t* sp1  = (bf16_t*)(ws + OSP1);
    bf16_t* sp2  = (bf16_t*)(ws + OSP2);
    bf16_t* w1T  = (bf16_t*)(ws + OW1T);
    bf16_t* w2T  = (bf16_t*)(ws + OW2T);
    float*  q0   = (float*)(ws + OQ0);
    float*  k0   = (float*)(ws + OK0);
    float*  qL   = (float*)(ws + OQL);
    float*  kL   = (float*)(ws + OKL);
    bf16_t* qbf  = (bf16_t*)(ws + OQB);
    bf16_t* kbf  = (bf16_t*)(ws + OKB);
    bf16_t* vbf  = (bf16_t*)(ws + OVB);
    bf16_t* kvT  = (bf16_t*)(ws + OKV);
    float*  km   = (float*)(ws + OKM);
    float*  zz   = (float*)(ws + OZ);
    float*  spin = (float*)(ws + OSPIN);
    float*  cbuf = (float*)(ws + OC);
    bf16_t* ubf  = (bf16_t*)(ws + OU);

    // ---- weight conversion (bf16, transposed to [N,K]) ----
    trbf_kernel<<<(1024 * 512) / 256, 256, 0, stream>>>(qk_w, qkT, 512, 1024);
    trbf_kernel<<<(512  * 512) / 256, 256, 0, stream>>>(v_w,  vT,  512, 512);
    spw_kernel <<<(1024 * 512) / 256, 256, 0, stream>>>(sp_cw, sp0, sp1, sp2);
    trbf_kernel<<<(2048 * 512) / 256, 256, 0, stream>>>(w1, w1T, 512, 2048);
    trbf_kernel<<<(512 * 2048) / 256, 256, 0, stream>>>(w2, w2T, 2048, 512);

    // ---- Block 1: linear attention with LePE ----
    ln_kernel<<<MROWS / 8, 256, 0, stream>>>(x, g1, b1, Abf, nullptr);
    gemm_bf16_kernel<0, true><<<dim3(4, MROWS / 128), 256, 0, stream>>>(
        Abf, qkT, qk_b, MROWS, 1024, 512, 0, q0, k0, nullptr);
    gemm_bf16_kernel<1, true><<<dim3(2, MROWS / 128), 256, 0, stream>>>(
        Abf, vT, v_b, MROWS, 512, 512, 0, nullptr, nullptr, vbf);
    lepe_kernel<<<(MROWS * 512) / 256, 256, 0, stream>>>(q0, lepe_w, lepe_b, qL, qbf);
    lepe_kernel<<<(MROWS * 512) / 256, 256, 0, stream>>>(k0, lepe_w, lepe_b, kL, kbf);
    kmean_kernel<<<(BN * 512) / 256, 256, 0, stream>>>(kL, km);
    z_kernel<<<(MROWS * HN) / 256, 256, 0, stream>>>(qL, km, zz);
    kv_kernel<<<BN * HN, 256, 0, stream>>>(kbf, vbf, kvT);
    attn_out_kernel<<<dim3(BN * HN, LN / 512), 256, 0, stream>>>(qbf, kvT, zz, x, out);

    // ---- Block 2: gated conv state path ----
    ln_kernel<<<MROWS / 8, 256, 0, stream>>>(out, g2, b2, nullptr, spin);   // sp_in
    ln_kernel<<<MROWS / 8, 256, 0, stream>>>(spin, spn_g, spn_b, Abf, nullptr);
    gemm_bf16_kernel<2, false><<<dim3(4, MROWS / 128), 256, 0, stream>>>(
        Abf, sp0, sp_cb, MROWS, 1024, 512, -1, cbuf, nullptr, nullptr);
    gemm_bf16_kernel<3, false><<<dim3(4, MROWS / 128), 256, 0, stream>>>(
        Abf, sp1, nullptr, MROWS, 1024, 512, 0, cbuf, nullptr, nullptr);
    gemm_bf16_kernel<3, false><<<dim3(4, MROWS / 128), 256, 0, stream>>>(
        Abf, sp2, nullptr, MROWS, 1024, 512, +1, cbuf, nullptr, nullptr);
    gate_kernel<<<(MROWS * 512) / 256, 256, 0, stream>>>(cbuf, spin, out);

    // ---- Block 3: FFN (exact GELU) ----
    ln_kernel<<<MROWS / 8, 256, 0, stream>>>(out, g3, b3, Abf, nullptr);
    gemm_bf16_kernel<4, true><<<dim3(8, MROWS / 128), 256, 0, stream>>>(
        Abf, w1T, bb1, MROWS, 2048, 512, 0, nullptr, nullptr, ubf);
    gemm_bf16_kernel<5, true><<<dim3(2, MROWS / 128), 256, 0, stream>>>(
        ubf, w2T, bb2, MROWS, 512, 2048, 0, out, nullptr, nullptr);

    (void)in_sizes; (void)n_in; (void)out_size; (void)ws_size;
}
